// SeqTranslatorRNN_15341623182092
// MI455X (gfx1250) — matrix operand order repaired and checked by reading, not compiled, not run on hardware
//
#include <hip/hip_runtime.h>
#include <hip/hip_bf16.h>

// ---------------------------------------------------------------------------
// SeqTranslatorRNN for MI455X (gfx1250, wave32, WMMA bf16)
//   B=64, T=512, C_IN=C_OUT=64, H=512
//  - weights packed once per call into bf16 WMMA B-fragment layout
//  - per recurrent step: one gru_cell kernel per layer (8 blocks x 256 thr);
//    each wave owns two 16x16 h-tiles, 4 accumulators (r, z, n_x, n_h),
//    v_wmma_f32_16x16x32_bf16 with fully-unrolled K loops (INK templated)
//    so all fragment loads use immediate offsets from one base pointer
//  - activations staged global->LDS via GLOBAL_LOAD_ASYNC_TO_LDS_B128
//    (ASYNCcnt) when the builtin exists, else plain uint4 copies
// ---------------------------------------------------------------------------

typedef __attribute__((ext_vector_type(16))) __bf16 v16bf;
typedef __attribute__((ext_vector_type(8)))  float  v8f;

#if defined(__has_builtin)
#if __has_builtin(__builtin_amdgcn_global_load_async_to_lds_b128) && \
    __has_builtin(__builtin_amdgcn_s_wait_asynccnt)
#define HAVE_ASYNC_LDS 1
#endif
#endif
#ifndef HAVE_ASYNC_LDS
#define HAVE_ASYNC_LDS 0
#endif

#if HAVE_ASYNC_LDS
typedef int v4i_vs __attribute__((vector_size(16)));
typedef __attribute__((address_space(1))) v4i_vs gv4i;   // global int4
typedef __attribute__((address_space(3))) v4i_vs lv4i;   // LDS int4
#endif

union FragU { v16bf v; uint4 u[2]; };

// B fragment: 32x16 (KxN) bf16, lane L holds 32 contiguous bytes.
// lane<16 : col n=lane,    K = kt*32 + 0..15 (dword v -> K=2v,2v+1)
// lane>=16: col n=lane-16, K = kt*32 + 16..31
__device__ __forceinline__ v16bf load_b_frag(const uint4* q) {
  FragU f; f.u[0] = q[0]; f.u[1] = q[1];
  return f.v;
}

// A fragment: 16x32 (MxK) bf16 in contiguous K order (matches packed B):
// lane<16 : row=mt*16+lane,    K = k0+0..15
// lane>=16: row=mt*16+lane-16, K = k0+16..31
__device__ __forceinline__ v16bf load_a_frag(const __bf16* s, int mt, int k0,
                                             int stride, int lane) {
  const int row  = mt * 16 + (lane & 15);
  const int koff = (lane < 16) ? 0 : 16;
  const __bf16* p = s + row * stride + k0 + koff;
  FragU f;
  f.u[0] = *(const uint4*)(p);
  f.u[1] = *(const uint4*)(p + 8);
  return f.v;
}

__device__ __forceinline__ v8f wmma_bf16(v16bf a, v16bf b, v8f c) {
  return __builtin_amdgcn_wmma_f32_16x16x32_bf16(false, a, false, b,
                                                 (short)0, c, false, false);
}

__device__ __forceinline__ float sigmoidf_fast(float x) {
  return 1.0f / (1.0f + __expf(-x));
}

// Copy 16 bytes global -> LDS (async path uses GLOBAL_LOAD_ASYNC_TO_LDS_B128).
__device__ __forceinline__ void stage16(__bf16* dstLds, const __bf16* srcGlob) {
#if HAVE_ASYNC_LDS
  __builtin_amdgcn_global_load_async_to_lds_b128(
      (gv4i*)(srcGlob), (lv4i*)(dstLds), 0, 0);
#else
  *(uint4*)dstLds = *(const uint4*)srcGlob;
#endif
}

__device__ __forceinline__ void stage_fence() {
#if HAVE_ASYNC_LDS
  __builtin_amdgcn_s_wait_asynccnt(0);
#endif
  __syncthreads();
}

// ---------------------------------------------------------------------------
// Pack fp32 weight W[R][C] (row-major) into bf16 WMMA B-fragments (RNE).
// ---------------------------------------------------------------------------
__global__ void pack_weights_kernel(const float* __restrict__ W,
                                    unsigned* __restrict__ dst, int R, int C) {
  const int ktiles = C >> 5;
  const int total  = (R >> 4) * ktiles * 256;  // dwords
  for (int i = blockIdx.x * blockDim.x + threadIdx.x; i < total;
       i += gridDim.x * blockDim.x) {
    const int f    = i >> 8;
    const int rr   = i & 255;
    const int lane = rr >> 3;
    const int v    = rr & 7;
    const int nt   = f / ktiles;
    const int kt   = f - nt * ktiles;
    const int n    = nt * 16 + (lane & 15);
    const int k    = kt * 32 + ((lane < 16) ? 0 : 16) + (v << 1);
    unsigned ua = __float_as_uint(W[n * C + k]);
    unsigned ub = __float_as_uint(W[n * C + k + 1]);
    ua += 0x7fffu + ((ua >> 16) & 1u);   // round-to-nearest-even bf16
    ub += 0x7fffu + ((ub >> 16) & 1u);
    dst[i] = (ua >> 16) | (ub & 0xffff0000u);
  }
}

// ---------------------------------------------------------------------------
// Zero initial hidden states, seed decoder frame from frame_0.
// ---------------------------------------------------------------------------
__global__ void init_state_kernel(float* __restrict__ h0F, float* __restrict__ h1F,
                                  __bf16* __restrict__ h0B, __bf16* __restrict__ h1B,
                                  const float* __restrict__ frame0,
                                  __bf16* __restrict__ frameB) {
  const int i = blockIdx.x * blockDim.x + threadIdx.x;
  if (i < 64 * 512) {
    h0F[i] = 0.0f; h1F[i] = 0.0f;
    h0B[i] = (__bf16)0.0f; h1B[i] = (__bf16)0.0f;
  }
  if (i < 64 * 64) frameB[i] = (__bf16)frame0[i];
}

// ---------------------------------------------------------------------------
// One GRU cell step for one layer. INK = input feature count (64 or 512).
// grid = 8 blocks (each owns 64 h-columns), block = 256 threads = 8 waves.
// Each wave computes two 16x16 h-tiles:
//   accR = x@Wr + h@Ur, accZ = x@Wz + h@Uz, accI = x@Wn, accH = h@Un
// x: strided fp32 x[:,:,t] (encoder L0) or row-major bf16 (64 x INK).
// ---------------------------------------------------------------------------
template <int INK>
__global__ __launch_bounds__(256) void gru_cell_kernel(
    const float* __restrict__ xStrided, int t, int T,
    const __bf16* __restrict__ xRow,
    const uint4* __restrict__ WihP, const uint4* __restrict__ WhhP,
    const float* __restrict__ bih, const float* __restrict__ bhh,
    const float* __restrict__ hInF, const __bf16* __restrict__ hInB,
    float* __restrict__ hOutF, __bf16* __restrict__ hOutB) {
  __shared__ __align__(16) __bf16 sAct[64 * 128];  // activation chunk (<=16KB)

  const int tid  = threadIdx.x;
  const int wave = tid >> 5;
  const int lane = tid & 31;

  // two tiles per wave: tile = wave*2+u -> mt = tile&3, jt = tile>>2
  const int mt0 = (wave * 2) & 3,     jt0 = (wave * 2) >> 2;
  const int mt1 = (wave * 2 + 1) & 3, jt1 = (wave * 2 + 1) >> 2;
  const int ntR0 = blockIdx.x * 4 + jt0;   // gate-r n-tile; z = +32, n = +64
  const int ntR1 = blockIdx.x * 4 + jt1;

  v8f accR[2] = {};
  v8f accZ[2] = {};
  v8f accI[2] = {};  // x-part of n-gate
  v8f accH[2] = {};  // h-part of n-gate (scaled by r in epilogue)

  // ---- phase 0: x @ Wih^T --------------------------------------------------
  {
    constexpr int KT = INK / 32;               // k-tiles in Wih
    constexpr int CH = (INK < 128) ? INK : 128;
    // per-tile gate-r base pointer (z at +32*KT frags, n at +64*KT frags)
    const uint4* wb[2] = {WihP + (size_t)ntR0 * KT * 64 + lane * 2,
                          WihP + (size_t)ntR1 * KT * 64 + lane * 2};
#pragma unroll
    for (int kc = 0; kc < INK; kc += CH) {
      __syncthreads();
      if (INK == 64 && xStrided) {
        // strided fp32 -> bf16 conversion (encoder layer 0 input x[:, :, t])
        for (int i = tid; i < 64 * CH; i += 256) {
          const int row = i / CH, c = i - row * CH;
          sAct[row * CH + c] = (__bf16)xStrided[(row * INK + kc + c) * T + t];
        }
        __syncthreads();
      } else {
        constexpr int C8 = CH >> 3;
        for (int i = tid; i < 64 * C8; i += 256) {
          const int row = i / C8, c8 = (i - row * C8) << 3;
          stage16(&sAct[row * CH + c8], &xRow[row * INK + kc + c8]);
        }
        stage_fence();
      }
#pragma unroll
      for (int u = 0; u < 2; ++u) {
        const int mt = u ? mt1 : mt0;
#pragma unroll
        for (int ks = 0; ks < CH / 32; ++ks) {
          const v16bf a = load_a_frag(sAct, mt, ks * 32, CH, lane);
          const uint4* q = wb[u] + ((kc >> 5) + ks) * 64;
          accR[u] = wmma_bf16(a, load_b_frag(q), accR[u]);
          accZ[u] = wmma_bf16(a, load_b_frag(q + 32 * KT * 64), accZ[u]);
          accI[u] = wmma_bf16(a, load_b_frag(q + 64 * KT * 64), accI[u]);
        }
      }
    }
  }

  // ---- phase 1: h @ Whh^T (K = 512, 4 chunks of 128) -----------------------
  {
    constexpr int KT = 16;
    const uint4* wb[2] = {WhhP + (size_t)ntR0 * KT * 64 + lane * 2,
                          WhhP + (size_t)ntR1 * KT * 64 + lane * 2};
#pragma unroll
    for (int kc = 0; kc < 512; kc += 128) {
      __syncthreads();
      for (int i = tid; i < 64 * 16; i += 256) {   // 64 rows x 16 uint4
        const int row = i >> 4, c8 = (i & 15) << 3;
        stage16(&sAct[row * 128 + c8], &hInB[row * 512 + kc + c8]);
      }
      stage_fence();
#pragma unroll
      for (int u = 0; u < 2; ++u) {
        const int mt = u ? mt1 : mt0;
#pragma unroll
        for (int ks = 0; ks < 4; ++ks) {
          const v16bf a = load_a_frag(sAct, mt, ks * 32, 128, lane);
          const uint4* q = wb[u] + ((kc >> 5) + ks) * 64;
          accR[u] = wmma_bf16(a, load_b_frag(q), accR[u]);
          accZ[u] = wmma_bf16(a, load_b_frag(q + 32 * KT * 64), accZ[u]);
          accH[u] = wmma_bf16(a, load_b_frag(q + 64 * KT * 64), accH[u]);
        }
      }
    }
  }

  // ---- epilogue: GRU nonlinearity, write h_new (fp32 + bf16) ---------------
#pragma unroll
  for (int u = 0; u < 2; ++u) {
    const int mt = u ? mt1 : mt0;
    const int jt = u ? jt1 : jt0;
    const int ncol = blockIdx.x * 64 + jt * 16 + (lane & 15);
    const float bR = bih[ncol]       + bhh[ncol];
    const float bZ = bih[512 + ncol] + bhh[512 + ncol];
    const float bI = bih[1024 + ncol];
    const float bH = bhh[1024 + ncol];
    const int mbase = mt * 16 + ((lane >> 4) << 3);
    const float*  hi = hInF  + mbase * 512 + ncol;
    float*        ho = hOutF + mbase * 512 + ncol;
    __bf16*       hb = hOutB + mbase * 512 + ncol;
#pragma unroll
    for (int r = 0; r < 8; ++r) {
      const float rg = sigmoidf_fast(accR[u][r] + bR);
      const float zg = sigmoidf_fast(accZ[u][r] + bZ);
      const float ng = tanhf(accI[u][r] + bI + rg * (accH[u][r] + bH));
      const float hn = (1.0f - zg) * ng + zg * hi[r * 512];
      ho[r * 512] = hn;
      hb[r * 512] = (__bf16)hn;
    }
  }
}

// ---------------------------------------------------------------------------
// Decoder FC: out = h1 @ fc_W^T + fc_b (64x64). Writes d_out[:, :, t] (fp32)
// and the next autoregressive frame (bf16). grid = 1 block, 256 threads.
// ---------------------------------------------------------------------------
__global__ __launch_bounds__(256) void fc_kernel(
    const __bf16* __restrict__ hB, const uint4* __restrict__ fcP,
    const float* __restrict__ fcBias, float* __restrict__ out,
    int t, int T, __bf16* __restrict__ frameOut) {
  __shared__ __align__(16) __bf16 sAct[64 * 128];
  const int tid = threadIdx.x, wave = tid >> 5, lane = tid & 31;
  const int mt0 = (wave * 2) & 3,     nt0 = (wave * 2) >> 2;
  const int mt1 = (wave * 2 + 1) & 3, nt1 = (wave * 2 + 1) >> 2;
  const uint4* wb[2] = {fcP + (size_t)nt0 * 16 * 64 + lane * 2,
                        fcP + (size_t)nt1 * 16 * 64 + lane * 2};
  v8f acc[2] = {};
#pragma unroll
  for (int kc = 0; kc < 512; kc += 128) {
    __syncthreads();
    for (int i = tid; i < 64 * 16; i += 256) {
      const int row = i >> 4, c8 = (i & 15) << 3;
      stage16(&sAct[row * 128 + c8], &hB[row * 512 + kc + c8]);
    }
    stage_fence();
#pragma unroll
    for (int u = 0; u < 2; ++u) {
      const int mt = u ? mt1 : mt0;
#pragma unroll
      for (int ks = 0; ks < 4; ++ks) {
        const v16bf a = load_a_frag(sAct, mt, ks * 32, 128, lane);
        acc[u] = wmma_bf16(a, load_b_frag(wb[u] + ((kc >> 5) + ks) * 64), acc[u]);
      }
    }
  }
#pragma unroll
  for (int u = 0; u < 2; ++u) {
    const int mt = u ? mt1 : mt0;
    const int nt = u ? nt1 : nt0;
    const int ncol = nt * 16 + (lane & 15);
    const float bb = fcBias[ncol];
    const int mbase = mt * 16 + ((lane >> 4) << 3);
#pragma unroll
    for (int r = 0; r < 8; ++r) {
      const int m = mbase + r;
      const float v = acc[u][r] + bb;
      out[(m * 64 + ncol) * T + t] = v;          // (B, C_OUT, T)
      frameOut[m * 64 + ncol] = (__bf16)v;
    }
  }
}

// ---------------------------------------------------------------------------
// Host orchestration.
// ---------------------------------------------------------------------------
extern "C" void kernel_launch(void* const* d_in, const int* in_sizes, int n_in,
                              void* d_out, int out_size, void* d_ws, size_t ws_size,
                              hipStream_t stream) {
  constexpr int T = 512, H = 512, B = 64, CIN = 64, COUT = 64;
  (void)in_sizes; (void)n_in; (void)out_size; (void)ws_size;

  const float* x      = (const float*)d_in[0];
  const float* frame0 = (const float*)d_in[1];
  const float* eWih0  = (const float*)d_in[2];
  const float* eWhh0  = (const float*)d_in[3];
  const float* ebih0  = (const float*)d_in[4];
  const float* ebhh0  = (const float*)d_in[5];
  const float* eWih1  = (const float*)d_in[6];
  const float* eWhh1  = (const float*)d_in[7];
  const float* ebih1  = (const float*)d_in[8];
  const float* ebhh1  = (const float*)d_in[9];
  const float* dWih0  = (const float*)d_in[10];
  const float* dWhh0  = (const float*)d_in[11];
  const float* dbih0  = (const float*)d_in[12];
  const float* dbhh0  = (const float*)d_in[13];
  const float* dWih1  = (const float*)d_in[14];
  const float* dWhh1  = (const float*)d_in[15];
  const float* dbih1  = (const float*)d_in[16];
  const float* dbhh1  = (const float*)d_in[17];
  const float* fcW    = (const float*)d_in[18];
  const float* fcB    = (const float*)d_in[19];
  float* dout = (float*)d_out;

  char* ws = (char*)d_ws;
  size_t off = 0;
  auto alloc = [&](size_t bytes) -> char* {
    char* p = ws + off;
    off = (off + bytes + 255) & ~(size_t)255;
    return p;
  };

  // packed bf16 weights
  uint4* eWih0P = (uint4*)alloc((size_t)3 * H * CIN * 2);
  uint4* eWhh0P = (uint4*)alloc((size_t)3 * H * H * 2);
  uint4* eWih1P = (uint4*)alloc((size_t)3 * H * H * 2);
  uint4* eWhh1P = (uint4*)alloc((size_t)3 * H * H * 2);
  uint4* dWih0P = (uint4*)alloc((size_t)3 * H * COUT * 2);
  uint4* dWhh0P = (uint4*)alloc((size_t)3 * H * H * 2);
  uint4* dWih1P = (uint4*)alloc((size_t)3 * H * H * 2);
  uint4* dWhh1P = (uint4*)alloc((size_t)3 * H * H * 2);
  uint4* fcP    = (uint4*)alloc((size_t)COUT * H * 2);
  // ping-pong hidden state (layer0 / layer1), fp32 + bf16, plus frame
  float*  h0F[2] = {(float*)alloc(B * H * 4), (float*)alloc(B * H * 4)};
  float*  h1F[2] = {(float*)alloc(B * H * 4), (float*)alloc(B * H * 4)};
  __bf16* h0B[2] = {(__bf16*)alloc(B * H * 2), (__bf16*)alloc(B * H * 2)};
  __bf16* h1B[2] = {(__bf16*)alloc(B * H * 2), (__bf16*)alloc(B * H * 2)};
  __bf16* frB[2] = {(__bf16*)alloc(B * COUT * 2), (__bf16*)alloc(B * COUT * 2)};

  auto pack = [&](const float* W, uint4* dst, int R, int C) {
    pack_weights_kernel<<<dim3(256), dim3(256), 0, stream>>>(W, (unsigned*)dst, R, C);
  };
  pack(eWih0, eWih0P, 3 * H, CIN);
  pack(eWhh0, eWhh0P, 3 * H, H);
  pack(eWih1, eWih1P, 3 * H, H);
  pack(eWhh1, eWhh1P, 3 * H, H);
  pack(dWih0, dWih0P, 3 * H, COUT);
  pack(dWhh0, dWhh0P, 3 * H, H);
  pack(dWih1, dWih1P, 3 * H, H);
  pack(dWhh1, dWhh1P, 3 * H, H);
  pack(fcW, fcP, COUT, H);

  init_state_kernel<<<dim3(128), dim3(256), 0, stream>>>(
      h0F[0], h1F[0], h0B[0], h1B[0], frame0, frB[0]);

  // --- encoder: 512 steps, two stacked GRU layers ---
  for (int t = 0; t < T; ++t) {
    const int pi = t & 1, po = pi ^ 1;
    gru_cell_kernel<64><<<dim3(8), dim3(256), 0, stream>>>(
        x, t, T, (const __bf16*)nullptr,
        eWih0P, eWhh0P, ebih0, ebhh0,
        h0F[pi], h0B[pi], h0F[po], h0B[po]);
    gru_cell_kernel<512><<<dim3(8), dim3(256), 0, stream>>>(
        (const float*)nullptr, 0, 0, h0B[po],
        eWih1P, eWhh1P, ebih1, ebhh1,
        h1F[pi], h1B[pi], h1F[po], h1B[po]);
  }

  // --- decoder: 512 autoregressive steps (states continue from encoder) ---
  for (int t = 0; t < T; ++t) {
    const int pi = t & 1, po = pi ^ 1;
    gru_cell_kernel<64><<<dim3(8), dim3(256), 0, stream>>>(
        (const float*)nullptr, 0, 0, frB[pi],
        dWih0P, dWhh0P, dbih0, dbhh0,
        h0F[pi], h0B[pi], h0F[po], h0B[po]);
    gru_cell_kernel<512><<<dim3(8), dim3(256), 0, stream>>>(
        (const float*)nullptr, 0, 0, h0B[po],
        dWih1P, dWhh1P, dbih1, dbhh1,
        h1F[pi], h1B[pi], h1F[po], h1B[po]);
    fc_kernel<<<dim3(1), dim3(256), 0, stream>>>(
        h1B[po], fcP, fcB, dout, t, T, frB[po]);
  }
}
